// MHA_27711128994546
// MI455X (gfx1250) — matrix operand order repaired and checked by reading, not compiled, not run on hardware
//
#include <hip/hip_runtime.h>

// ---------------------------------------------------------------------------
// out = softmax( relu(qW1^T+b1) @ relu(kW2^T+b2)^T * 1/sqrt(3) ) @ v
// B=4, Nq=2048, Nf=16384, D=3, C=768.  gfx1250 (CDNA5, wave32, WMMA).
// Fast path: producer/consumer pipelined flash attention, f16 V in ws.
// ---------------------------------------------------------------------------

#define B_     4
#define NQ_    2048
#define NF_    16384
#define C_     768
#define CW_    96      // C columns per consumer wave (8*96 = 768)
#define SCALE_ 0.57735026918962576451f  // 1/sqrt(3)

#define PSTRF_ 72      // fast: P row stride in halves (144B, 16B-aligned)
#define KTL_   32      // fallback: keys per iteration
#define VSTR_  40      // fallback: LDS half-stride for transposed V rows
#define PSTRL_ 40      // fallback: LDS half-stride for P tiles

typedef __attribute__((ext_vector_type(2)))  float    v2f;
typedef __attribute__((ext_vector_type(8)))  float    v8f;
typedef __attribute__((ext_vector_type(2)))  _Float16 v2h;
typedef __attribute__((ext_vector_type(4)))  _Float16 v4h;
typedef __attribute__((ext_vector_type(8)))  _Float16 v8h;
typedef __attribute__((ext_vector_type(16))) _Float16 v16h;

// --------------------------- lane-row reductions ---------------------------
__device__ __forceinline__ float pl16(float x, int s0, int s1) {
    int xi = __builtin_bit_cast(int, x);
    int r  = __builtin_amdgcn_permlane16(xi, xi, s0, s1, false, false);
    return __builtin_bit_cast(float, r);
}
__device__ __forceinline__ float rowmax16(float x) {
    x = fmaxf(x, pl16(x, (int)0x67452301, (int)0xEFCDAB89));  // xor 1
    x = fmaxf(x, pl16(x, (int)0x54761032, (int)0xDCFE98BA));  // xor 2
    x = fmaxf(x, pl16(x, (int)0x32107654, (int)0xBA98FEDC));  // xor 4
    x = fmaxf(x, pl16(x, (int)0xFEDCBA98, (int)0x76543210));  // xor 8
    return x;
}
__device__ __forceinline__ float rowsum16(float x) {
    x += pl16(x, (int)0x67452301, (int)0xEFCDAB89);
    x += pl16(x, (int)0x54761032, (int)0xDCFE98BA);
    x += pl16(x, (int)0x32107654, (int)0xBA98FEDC);
    x += pl16(x, (int)0xFEDCBA98, (int)0x76543210);
    return x;
}

// ---------------------------------------------------------------------------
// Projection: out4[i] = { relu(x W^T + b) * postscale, 0 }
// ---------------------------------------------------------------------------
__global__ __launch_bounds__(256) void proj3_kernel(
    const float* __restrict__ x, const float* __restrict__ W,
    const float* __restrict__ bias, float* __restrict__ out4,
    int npts, float postscale)
{
    int i = blockIdx.x * 256 + threadIdx.x;
    if (i >= npts) return;
    float x0 = x[3*i + 0], x1 = x[3*i + 1], x2 = x[3*i + 2];
    float4 o;
    o.x = fmaxf(0.f, x0*W[0] + x1*W[1] + x2*W[2] + bias[0]) * postscale;
    o.y = fmaxf(0.f, x0*W[3] + x1*W[4] + x2*W[5] + bias[1]) * postscale;
    o.z = fmaxf(0.f, x0*W[6] + x1*W[7] + x2*W[8] + bias[2]) * postscale;
    o.w = 0.f;
    ((float4*)out4)[i] = o;
}

// ---------------------------------------------------------------------------
// V pre-pass: f32 [B][NF][C] -> f16 [B][C][NF]; each 64-key group permuted so
// position j holds original key (j%4)*16 + j/4 (matches producer's b64 pack).
// ---------------------------------------------------------------------------
__global__ __launch_bounds__(256) void vt_transpose_kernel(
    const float* __restrict__ v, _Float16* __restrict__ vt)
{
    __shared__ _Float16 tile[64 * 72];

    const int t      = threadIdx.x;
    const int nchk   = C_ / 64;                       // 12
    const int cchunk = blockIdx.x % nchk;
    const int g      = (blockIdx.x / nchk) % (NF_ / 64);
    const int b      = blockIdx.x / (nchk * (NF_ / 64));

    const float* src = v + ((size_t)b * NF_ + (size_t)g * 64) * C_ + cchunk * 64;

    #pragma unroll
    for (int i = 0; i < 8; ++i) {
        int idx = i * 256 + t;
        int kk  = idx >> 5;
        int c2  = (idx & 31) << 1;
        const float* p = src + (size_t)kk * C_ + c2;
        v2h h; h.x = (_Float16)p[0]; h.y = (_Float16)p[1];
        *(v2h*)(&tile[kk * 72 + c2]) = h;
    }
    __syncthreads();

    const int c_local = t >> 2;
    const int q       = t & 3;
    _Float16 hbuf[16];
    #pragma unroll
    for (int jj = 0; jj < 16; ++jj) {
        int orig = (jj & 3) * 16 + q * 4 + (jj >> 2);
        hbuf[jj] = tile[orig * 72 + c_local];
    }
    _Float16* dst = vt + ((size_t)b * C_ + cchunk * 64 + c_local) * NF_
                       + (size_t)g * 64 + q * 16;
    v8h o0, o1;
    #pragma unroll
    for (int jj = 0; jj < 8; ++jj) { o0[jj] = hbuf[jj]; o1[jj] = hbuf[8 + jj]; }
    *(v8h*)(dst)     = o0;
    *(v8h*)(dst + 8) = o1;
}

// ---------------------------------------------------------------------------
// FAST main kernel.  288 threads = 9 waves.
//   waves 0..7 : consumers, 96 C-columns each, 12x wmma f16 / iter
//   wave  8    : softmax producer (runs one 64-key tile AHEAD),
//                4x wmma f32 16x16x4 + permlane16 maxes + exps +
//                2x ones-B wmma f16 for the running row-sum.
// Double-buffered P/alpha in LDS; one __syncthreads per iteration.
// Producer publishes a "rescale needed" flag so consumers skip the alpha
// multiplies (alpha == 1.0 exactly once the running max has stabilized).
// ---------------------------------------------------------------------------
__global__ __launch_bounds__(288) void flash_attn_fast_kernel(
    const _Float16* __restrict__ vt,   // [B][C][NF] f16 (64-key permuted)
    const float* __restrict__ qp4,     // [B][NQ][4] f32 (pre-scaled)
    const float* __restrict__ kp4,     // [B][NF][4] f32
    float* __restrict__ out)           // [B][NQ][C] f32
{
    __shared__ __align__(16) _Float16 pbuf[2][16 * PSTRF_];  // P tiles (16x64)
    __shared__ __align__(16) float    ashare[2][20];         // alpha[16] + flag
    __shared__ __align__(16) float    lshare[16];            // final row sums

    const int tid  = threadIdx.x;
    const int lane = tid & 31;
    const int wave = tid >> 5;
    const int hi   = lane >> 4;
    const int n    = lane & 15;

    const int b     = blockIdx.x / (NQ_ / 16);
    const int qbase = (blockIdx.x % (NQ_ / 16)) * 16;
    const int NIT   = NF_ / 64;

    if (wave == 8) {
        // =================== PRODUCER ===================
        const float* kpb = kp4 + (size_t)b * NF_ * 4;
        const float* qprow = qp4 + ((size_t)(b * NQ_ + qbase + n)) * 4 + 2 * hi;
        v2f aq; aq.x = qprow[0]; aq.y = qprow[1];

        v16h bones;
        #pragma unroll
        for (int i = 0; i < 16; ++i) bones[i] = (_Float16)1.0f;

        float mrow[8];
        v8f lacc, zc;
        #pragma unroll
        for (int r = 0; r < 8; ++r) { mrow[r] = -__builtin_inff(); lacc[r] = 0.f; zc[r] = 0.f; }

        auto produce = [&](int it2) {
            const int kbase = it2 * 64;
            const int buf   = it2 & 1;
            v8f S[4];
            #pragma unroll
            for (int t2 = 0; t2 < 4; ++t2) {
                const float* kr = kpb + (size_t)(kbase + t2 * 16 + n) * 4 + 2 * hi;
                v2f bk; bk.x = kr[0]; bk.y = kr[1];
                S[t2] = __builtin_amdgcn_wmma_f32_16x16x4_f32(
                            false, aq, false, bk, (short)0, zc, false, false);
            }
            float al[8];
            #pragma unroll
            for (int r = 0; r < 8; ++r) {
                float mt = fmaxf(fmaxf(S[0][r], S[1][r]), fmaxf(S[2][r], S[3][r]));
                mt = rowmax16(mt);
                float mnew = fmaxf(mrow[r], mt);
                al[r]   = __expf(mrow[r] - mnew);
                mrow[r] = mnew;
                v4h ph;                                  // positions n*4 + t
                ph.x = (_Float16)__expf(S[0][r] - mnew);
                ph.y = (_Float16)__expf(S[1][r] - mnew);
                ph.z = (_Float16)__expf(S[2][r] - mnew);
                ph.w = (_Float16)__expf(S[3][r] - mnew);
                *(v4h*)(&pbuf[buf][(r + 8 * hi) * PSTRF_ + n * 4]) = ph;
            }
            // "rescale needed" flag: any alpha != 1.0 across all 16 rows
            float dmax = 0.f;
            #pragma unroll
            for (int r = 0; r < 8; ++r) dmax = fmaxf(dmax, fabsf(al[r] - 1.0f));
            dmax = rowmax16(dmax);
            dmax = fmaxf(dmax, __shfl_xor(dmax, 16, 32));   // cross hi-halves
            // running row-sum via ones-B wmma (reads back own P, same wave)
            const _Float16* prow = &pbuf[buf][n * PSTRF_];
            v8h a0 = *(const v8h*)(prow +      8 * hi);
            v8h a1 = *(const v8h*)(prow + 16 + 8 * hi);
            v8h a2 = *(const v8h*)(prow + 32 + 8 * hi);
            v8h a3 = *(const v8h*)(prow + 48 + 8 * hi);
            v16h pa0 = __builtin_shufflevector(a0, a1, 0,1,2,3,4,5,6,7,8,9,10,11,12,13,14,15);
            v16h pa1 = __builtin_shufflevector(a2, a3, 0,1,2,3,4,5,6,7,8,9,10,11,12,13,14,15);
            #pragma unroll
            for (int r = 0; r < 8; ++r) lacc[r] *= al[r];
            lacc = __builtin_amdgcn_wmma_f32_16x16x32_f16(
                       false, pa0, false, bones, (short)0, lacc, false, false);
            lacc = __builtin_amdgcn_wmma_f32_16x16x32_f16(
                       false, pa1, false, bones, (short)0, lacc, false, false);
            // broadcast alpha + flag
            if (n == 0) {
                float4 f0 = make_float4(al[0], al[1], al[2], al[3]);
                float4 f1 = make_float4(al[4], al[5], al[6], al[7]);
                *(float4*)(&ashare[buf][8 * hi])     = f0;
                *(float4*)(&ashare[buf][8 * hi + 4]) = f1;
                if (hi == 0) ashare[buf][16] = (dmax > 0.f) ? 1.0f : 0.0f;
            }
        };

        produce(0);
        __syncthreads();
        for (int it = 0; it < NIT; ++it) {
            if (it + 1 < NIT) produce(it + 1);
            __syncthreads();
        }
        if (n == 0) {
            float4 l0 = make_float4(lacc[0], lacc[1], lacc[2], lacc[3]);
            float4 l1 = make_float4(lacc[4], lacc[5], lacc[6], lacc[7]);
            *(float4*)(&lshare[8 * hi])     = l0;
            *(float4*)(&lshare[8 * hi + 4]) = l1;
        }
        __syncthreads();
    } else {
        // =================== CONSUMERS ===================
        const int cbase = wave * CW_;
        const _Float16* vtb = vt + (size_t)b * C_ * NF_;

        v8f acc[6];
        #pragma unroll
        for (int j = 0; j < 6; ++j)
            #pragma unroll
            for (int r = 0; r < 8; ++r) acc[j][r] = 0.f;

        __syncthreads();
        for (int it = 0; it < NIT; ++it) {
            const int kbase = it * 64;
            const int buf   = it & 1;

            // P fragments
            const _Float16* prow = &pbuf[buf][n * PSTRF_];
            v8h a0 = *(const v8h*)(prow +      8 * hi);
            v8h a1 = *(const v8h*)(prow + 16 + 8 * hi);
            v8h a2 = *(const v8h*)(prow + 32 + 8 * hi);
            v8h a3 = *(const v8h*)(prow + 48 + 8 * hi);
            v16h pa0 = __builtin_shufflevector(a0, a1, 0,1,2,3,4,5,6,7,8,9,10,11,12,13,14,15);
            v16h pa1 = __builtin_shufflevector(a2, a3, 0,1,2,3,4,5,6,7,8,9,10,11,12,13,14,15);

            // K-half 0 B fragments: issue all 12 b128 loads up front
            v16h bv0[6];
            #pragma unroll
            for (int j = 0; j < 6; ++j) {
                const _Float16* vr = vtb + (size_t)(cbase + j * 16 + n) * NF_ + kbase;
                v8h b0 = *(const v8h*)(vr + 8 * hi);
                v8h b1 = *(const v8h*)(vr + 16 + 8 * hi);
                bv0[j] = __builtin_shufflevector(b0, b1, 0,1,2,3,4,5,6,7,8,9,10,11,12,13,14,15);
            }

            // alpha rescale, skipped via scalar branch when alpha == 1
            float flagf = ashare[buf][16];
            if (__builtin_amdgcn_readfirstlane(__builtin_bit_cast(int, flagf)) != 0) {
                float4 af0 = *(const float4*)(&ashare[buf][8 * hi]);
                float4 af1 = *(const float4*)(&ashare[buf][8 * hi + 4]);
                float al[8] = { af0.x, af0.y, af0.z, af0.w,
                                af1.x, af1.y, af1.z, af1.w };
                #pragma unroll
                for (int j = 0; j < 6; ++j)
                    #pragma unroll
                    for (int r = 0; r < 8; ++r) acc[j][r] *= al[r];
            }

            #pragma unroll
            for (int j = 0; j < 6; ++j)
                acc[j] = __builtin_amdgcn_wmma_f32_16x16x32_f16(
                             false, pa0, false, bv0[j], (short)0, acc[j], false, false);

            // K-half 1
            v16h bv1[6];
            #pragma unroll
            for (int j = 0; j < 6; ++j) {
                const _Float16* vr = vtb + (size_t)(cbase + j * 16 + n) * NF_ + kbase;
                v8h b2 = *(const v8h*)(vr + 32 + 8 * hi);
                v8h b3 = *(const v8h*)(vr + 48 + 8 * hi);
                bv1[j] = __builtin_shufflevector(b2, b3, 0,1,2,3,4,5,6,7,8,9,10,11,12,13,14,15);
            }
            #pragma unroll
            for (int j = 0; j < 6; ++j)
                acc[j] = __builtin_amdgcn_wmma_f32_16x16x32_f16(
                             false, pa1, false, bv1[j], (short)0, acc[j], false, false);

            __syncthreads();
        }
        __syncthreads();

        float4 lf0 = *(const float4*)(&lshare[8 * hi]);
        float4 lf1 = *(const float4*)(&lshare[8 * hi + 4]);
        float lr[8] = { lf0.x, lf0.y, lf0.z, lf0.w,
                        lf1.x, lf1.y, lf1.z, lf1.w };

        float* ob = out + ((size_t)b * NQ_ + qbase) * C_;
        #pragma unroll
        for (int r = 0; r < 8; ++r) {
            float inv = 1.0f / lr[r];
            int m = r + 8 * hi;
            #pragma unroll
            for (int j = 0; j < 6; ++j)
                ob[(size_t)m * C_ + cbase + j * 16 + n] = acc[j][r] * inv;
        }
    }
}

// ---------------------------------------------------------------------------
// FALLBACK main kernel (ws too small for f16 V copy): LDS-staged V tiles.
// ---------------------------------------------------------------------------
__global__ __launch_bounds__(256) void flash_attn_lds_kernel(
    const float* __restrict__ v, const float* __restrict__ qp4,
    const float* __restrict__ kp4, float* __restrict__ out)
{
    __shared__ _Float16 vtl[C_ * VSTR_];
    __shared__ _Float16 pbuf[8 * 16 * PSTRL_];

    const int tid  = threadIdx.x;
    const int lane = tid & 31;
    const int wave = tid >> 5;
    const int hi   = lane >> 4;
    const int n    = lane & 15;

    const int b     = blockIdx.x / (NQ_ / 16);
    const int qbase = (blockIdx.x % (NQ_ / 16)) * 16;
    const int cbase = wave * CW_;

    const float* vbat = v   + (size_t)b * NF_ * C_;
    const float* kpb  = kp4 + (size_t)b * NF_ * 4;
    _Float16*    mypb = pbuf + wave * 16 * PSTRL_;

    const float* qprow = qp4 + ((size_t)(b * NQ_ + qbase + n)) * 4 + 2 * hi;
    v2f aq; aq.x = qprow[0]; aq.y = qprow[1];

    float mrow[8], lrow[8];
    v8f acc[6], zc;
    #pragma unroll
    for (int r = 0; r < 8; ++r) { mrow[r] = -__builtin_inff(); lrow[r] = 0.f; zc[r] = 0.f; }
    #pragma unroll
    for (int j = 0; j < 6; ++j)
        #pragma unroll
        for (int r = 0; r < 8; ++r) acc[j][r] = 0.f;

    for (int kbase = 0; kbase < NF_; kbase += KTL_) {
        if (kbase + KTL_ < NF_) {
            const char* nxt = (const char*)(vbat + (size_t)(kbase + KTL_) * C_);
            __builtin_prefetch(nxt + (size_t)tid * 384, 0, 0);
        }
        __syncthreads();
        for (int i = tid; i < (KTL_ * C_) / 4; i += 256) {
            int kk = i / (C_ / 4);
            int c4 = (i % (C_ / 4)) * 4;
            float4 x = *(const float4*)(vbat + (size_t)(kbase + kk) * C_ + c4);
            vtl[(c4 + 0) * VSTR_ + kk] = (_Float16)x.x;
            vtl[(c4 + 1) * VSTR_ + kk] = (_Float16)x.y;
            vtl[(c4 + 2) * VSTR_ + kk] = (_Float16)x.z;
            vtl[(c4 + 3) * VSTR_ + kk] = (_Float16)x.w;
        }
        __syncthreads();

        const float* kr0 = kpb + (size_t)(kbase +      n) * 4 + 2 * hi;
        const float* kr1 = kpb + (size_t)(kbase + 16 + n) * 4 + 2 * hi;
        v2f bk0; bk0.x = kr0[0]; bk0.y = kr0[1];
        v2f bk1; bk1.x = kr1[0]; bk1.y = kr1[1];
        v8f S0 = __builtin_amdgcn_wmma_f32_16x16x4_f32(
                     false, aq, false, bk0, (short)0, zc, false, false);
        v8f S1 = __builtin_amdgcn_wmma_f32_16x16x4_f32(
                     false, aq, false, bk1, (short)0, zc, false, false);

        float alpha[8];
        #pragma unroll
        for (int r = 0; r < 8; ++r) {
            float mt = rowmax16(fmaxf(S0[r], S1[r]));
            float mnew = fmaxf(mrow[r], mt);
            float p0 = __expf(S0[r] - mnew);
            float p1 = __expf(S1[r] - mnew);
            float rs = rowsum16(p0 + p1);
            alpha[r] = __expf(mrow[r] - mnew);
            lrow[r]  = lrow[r] * alpha[r] + rs;
            mrow[r]  = mnew;
            mypb[(r + 8 * hi) * PSTRL_ +      n] = (_Float16)p0;
            mypb[(r + 8 * hi) * PSTRL_ + 16 + n] = (_Float16)p1;
        }

        const _Float16* prow = mypb + n * PSTRL_;
        v8h plo = *(const v8h*)(prow + 8 * hi);
        v8h phi = *(const v8h*)(prow + 16 + 8 * hi);
        v16h pa = __builtin_shufflevector(plo, phi, 0,1,2,3,4,5,6,7,8,9,10,11,12,13,14,15);

        #pragma unroll
        for (int j = 0; j < 6; ++j) {
            const _Float16* vrow = vtl + (cbase + j * 16 + n) * VSTR_ + 8 * hi;
            v8h b0 = *(const v8h*)(vrow);
            v8h b1 = *(const v8h*)(vrow + 16);
            v16h bv = __builtin_shufflevector(b0, b1, 0,1,2,3,4,5,6,7,8,9,10,11,12,13,14,15);
            v8f c0 = acc[j];
            #pragma unroll
            for (int r = 0; r < 8; ++r) c0[r] *= alpha[r];
            acc[j] = __builtin_amdgcn_wmma_f32_16x16x32_f16(
                         false, pa, false, bv, (short)0, c0, false, false);
        }
    }

    float* ob = out + ((size_t)b * NQ_ + qbase) * C_;
    #pragma unroll
    for (int r = 0; r < 8; ++r) {
        float inv = 1.0f / lrow[r];
        int m = r + 8 * hi;
        #pragma unroll
        for (int j = 0; j < 6; ++j)
            ob[(size_t)m * C_ + cbase + j * 16 + n] = acc[j][r] * inv;
    }
}

// ---------------------------------------------------------------------------
extern "C" void kernel_launch(void* const* d_in, const int* in_sizes, int n_in,
                              void* d_out, int out_size, void* d_ws, size_t ws_size,
                              hipStream_t stream)
{
    const float* q  = (const float*)d_in[0];
    const float* k  = (const float*)d_in[1];
    const float* v  = (const float*)d_in[2];
    const float* W1 = (const float*)d_in[3];
    const float* b1 = (const float*)d_in[4];
    const float* W2 = (const float*)d_in[5];
    const float* b2 = (const float*)d_in[6];
    float* out = (float*)d_out;

    const size_t vt_bytes = (size_t)B_ * C_ * NF_ * sizeof(_Float16);
    const size_t qp_bytes = (size_t)B_ * NQ_ * 4 * sizeof(float);
    const size_t kp_bytes = (size_t)B_ * NF_ * 4 * sizeof(float);

    const int nq = B_ * NQ_;
    const int nk = B_ * NF_;

    if (ws_size >= vt_bytes + qp_bytes + kp_bytes) {
        _Float16* vt16 = (_Float16*)d_ws;
        float* qp4 = (float*)((char*)d_ws + vt_bytes);
        float* kp4 = qp4 + (size_t)B_ * NQ_ * 4;

        proj3_kernel<<<(nq + 255) / 256, 256, 0, stream>>>(q, W1, b1, qp4, nq, SCALE_);
        proj3_kernel<<<(nk + 255) / 256, 256, 0, stream>>>(k, W2, b2, kp4, nk, 1.0f);
        vt_transpose_kernel<<<B_ * (NF_ / 64) * (C_ / 64), 256, 0, stream>>>(v, vt16);
        flash_attn_fast_kernel<<<B_ * (NQ_ / 16), 288, 0, stream>>>(vt16, qp4, kp4, out);
    } else {
        float* qp4 = (float*)d_ws;
        float* kp4 = qp4 + (size_t)B_ * NQ_ * 4;
        proj3_kernel<<<(nq + 255) / 256, 256, 0, stream>>>(q, W1, b1, qp4, nq, SCALE_);
        proj3_kernel<<<(nk + 255) / 256, 256, 0, stream>>>(k, W2, b2, kp4, nk, 1.0f);
        flash_attn_lds_kernel<<<B_ * (NQ_ / 16), 256, 0, stream>>>(v, qp4, kp4, out);
    }
}
